// RelateModule_26792005992468
// MI455X (gfx1250) — matrix operand order repaired and checked by reading, not compiled, not run on hardware
//
#include <hip/hip_runtime.h>

#define D_OBJ 4096
#define D_TXT 1024
#define N_OBJ 4096
#define M_ANCHOR 8
#define H_MLP 128
#define W1_ROW (2 * D_OBJ + 3)  // 8195

typedef __attribute__((ext_vector_type(16))) __bf16 v16bf;
typedef __attribute__((ext_vector_type(8)))  float  v8f;

// ---------------------------------------------------------------------------
// ws layout (float offsets):
//   [0      , 4096 )  anchor_rep
//   [4096   , 8192 )  query_vec
//   [8192   , 20480)  spatial_offset (N,3)
//   [20480  , 20608)  cvec (128) = b1 + W1[:, :4096] @ query
//   [20608  , 24704)  scores (N)
//   [24704  , 28800)  weights (N)
//   [28800  , 28800 + 262144)  W1bf: bf16 copy of W1[:, 4096:8192], 128x4096
// total ~1.16 MB
// ---------------------------------------------------------------------------

// Kernel A: anchor mean, per-column argmax -> spatial offsets
__global__ void prep_kernel(const float* __restrict__ anchor_feats,
                            const float* __restrict__ coords,
                            float* __restrict__ anchor_rep,
                            float* __restrict__ spatial) {
    int j = blockIdx.x * blockDim.x + threadIdx.x;
    if (j >= D_OBJ) return;
    float s = 0.f;
    float best = -3.402823466e38f;
    int bi = 0;
#pragma unroll
    for (int m = 0; m < M_ANCHOR; ++m) {
        float v = anchor_feats[m * D_OBJ + j];
        s += v;
        if (v > best) { best = v; bi = m; }   // strict > keeps first max (jnp.argmax)
    }
    anchor_rep[j] = s * (1.f / M_ANCHOR);
    float ax = coords[bi * 3 + 0], ay = coords[bi * 3 + 1], az = coords[bi * 3 + 2];
    spatial[j * 3 + 0] = coords[j * 3 + 0] - ax;
    spatial[j * 3 + 1] = coords[j * 3 + 1] - ay;
    spatial[j * 3 + 2] = coords[j * 3 + 2] - az;
}

// Kernel B: query = tanh(Wa@anchor_rep + ba + Wr@rel + br), one wave per row
__global__ void query_gemv(const float* __restrict__ Wa, const float* __restrict__ ba,
                           const float* __restrict__ Wr, const float* __restrict__ br,
                           const float* __restrict__ anchor_rep,
                           const float* __restrict__ rel,
                           float* __restrict__ query) {
    int wave = threadIdx.x >> 5, lane = threadIdx.x & 31;
    int r = blockIdx.x * (blockDim.x >> 5) + wave;
    if (r >= D_OBJ) return;
    const float4* wa = (const float4*)(Wa + (size_t)r * D_OBJ);
    const float4* ar = (const float4*)anchor_rep;
    float acc = 0.f;
    for (int i = lane; i < D_OBJ / 4; i += 32) {
        float4 a = wa[i], b = ar[i];
        acc += a.x * b.x + a.y * b.y + a.z * b.z + a.w * b.w;
    }
    const float4* wr = (const float4*)(Wr + (size_t)r * D_TXT);
    const float4* rt = (const float4*)rel;
    for (int i = lane; i < D_TXT / 4; i += 32) {
        float4 a = wr[i], b = rt[i];
        acc += a.x * b.x + a.y * b.y + a.z * b.z + a.w * b.w;
    }
#pragma unroll
    for (int m = 16; m >= 1; m >>= 1) acc += __shfl_xor(acc, m, 32);
    if (lane == 0) query[r] = tanhf(acc + ba[r] + br[r]);
}

// Kernel C: cvec[h] = b1[h] + dot(W1[h, 0:4096], query). Row stride 8195 is odd
// -> scalar loads (only 2 MB total).
__global__ void cvec_kernel(const float* __restrict__ W1, const float* __restrict__ b1,
                            const float* __restrict__ query, float* __restrict__ cvec) {
    int wave = threadIdx.x >> 5, lane = threadIdx.x & 31;
    int r = blockIdx.x * (blockDim.x >> 5) + wave;
    if (r >= H_MLP) return;
    const float* wrow = W1 + (size_t)r * W1_ROW;
    float acc = 0.f;
    for (int k = lane; k < D_OBJ; k += 32) acc += wrow[k] * query[k];
#pragma unroll
    for (int m = 16; m >= 1; m >>= 1) acc += __shfl_xor(acc, m, 32);
    if (lane == 0) cvec[r] = acc + b1[r];
}

// Kernel D: pre-convert W1 object block (columns 4096..8191) to bf16, stride 4096.
__global__ void w1cvt_kernel(const float* __restrict__ W1, __bf16* __restrict__ W1bf) {
    int t = blockIdx.x * blockDim.x + threadIdx.x;
    if (t >= H_MLP * D_OBJ) return;
    int h = t >> 12, k = t & (D_OBJ - 1);
    W1bf[t] = (__bf16)W1[(size_t)h * W1_ROW + D_OBJ + k];
}

// Kernel E: H = relu(obj @ W1_obj^T + cvec + spatial @ W1_sp^T); scores = H @ W2 + b2.
// One block = one 16-row M-tile; wave w owns N-tile (h = 16w..16w+15); hidden tile
// lives only in the v8f accumulator; scores reduced via shuffles + LDS.
__launch_bounds__(256)
__global__ void gemm_score_kernel(const float* __restrict__ obj,
                                  const __bf16* __restrict__ W1bf,
                                  const float* __restrict__ W1,
                                  const float* __restrict__ cvec,
                                  const float* __restrict__ spatial,
                                  const float* __restrict__ W2,
                                  const float* __restrict__ b2,
                                  float* __restrict__ scores) {
    const int wave = threadIdx.x >> 5;
    const int lane = threadIdx.x & 31;
    const int half = lane >> 4;      // K-half select per ISA fragment layout
    const int ncol = lane & 15;
    const int m0 = blockIdx.x * 16;
    const int h0 = wave * 16;
    const int row = m0 + ncol;       // A: lane indexes M row
    const int h   = h0 + ncol;       // B: lane indexes N column (hidden unit)

    // A lane layout: 16 contiguous K per lane; lanes16-31 start at K+16
    const float*  arow = obj  + (size_t)row * D_OBJ + 16 * half;
    // B lane layout: 16 contiguous K per lane; lanes16-31 start at K+16
    const __bf16* brow = W1bf + (size_t)h * D_OBJ + 16 * half;

    v8f acc = {0.f, 0.f, 0.f, 0.f, 0.f, 0.f, 0.f, 0.f};
#pragma unroll 4
    for (int k0 = 0; k0 < D_OBJ; k0 += 32) {
        float4 x0 = *(const float4*)(arow + k0 + 0);
        float4 x1 = *(const float4*)(arow + k0 + 4);
        float4 x2 = *(const float4*)(arow + k0 + 8);
        float4 x3 = *(const float4*)(arow + k0 + 12);
        v16bf a;
        a[0]  = (__bf16)x0.x; a[1]  = (__bf16)x0.y; a[2]  = (__bf16)x0.z; a[3]  = (__bf16)x0.w;
        a[4]  = (__bf16)x1.x; a[5]  = (__bf16)x1.y; a[6]  = (__bf16)x1.z; a[7]  = (__bf16)x1.w;
        a[8]  = (__bf16)x2.x; a[9]  = (__bf16)x2.y; a[10] = (__bf16)x2.z; a[11] = (__bf16)x2.w;
        a[12] = (__bf16)x3.x; a[13] = (__bf16)x3.y; a[14] = (__bf16)x3.z; a[15] = (__bf16)x3.w;
        v16bf b = *(const v16bf*)(brow + k0);   // 32B aligned
        acc = __builtin_amdgcn_wmma_f32_16x16x32_bf16(
            /*neg_a=*/false, a, /*neg_b=*/false, b,
            /*c_mod=*/(short)0, acc, /*reuse_a=*/false, /*reuse_b=*/false);
    }

    // Epilogue: D layout — VGPR v, lanes0-15: (M=v, N=lane); lanes16-31: (M=8+v, N=lane-16)
    float cval = cvec[h];
    float w2v  = W2[h];
    const float* wsp = W1 + (size_t)h * W1_ROW + 2 * D_OBJ;
    float wx = wsp[0], wy = wsp[1], wz = wsp[2];
    float p[8];
#pragma unroll
    for (int v = 0; v < 8; ++v) {
        int m = m0 + v + 8 * half;
        float sx = spatial[m * 3 + 0], sy = spatial[m * 3 + 1], sz = spatial[m * 3 + 2];
        float val = acc[v] + cval + sx * wx + sy * wy + sz * wz;
        val = fmaxf(val, 0.f);
        p[v] = val * w2v;
    }
    // butterfly over the 16 N-lanes of each half
#pragma unroll
    for (int mask = 1; mask <= 8; mask <<= 1) {
#pragma unroll
        for (int v = 0; v < 8; ++v) p[v] += __shfl_xor(p[v], mask, 32);
    }
    __shared__ float lds[8][16];
    if (ncol == 0) {
#pragma unroll
        for (int v = 0; v < 8; ++v) lds[wave][v + 8 * half] = p[v];
    }
    __syncthreads();
    if (threadIdx.x < 16) {
        float s = b2[0];
#pragma unroll
        for (int ww = 0; ww < 8; ++ww) s += lds[ww][threadIdx.x];
        scores[m0 + threadIdx.x] = s;
    }
}

// Kernel F: stable softmax over 4096 scores, single 1024-thread block.
__global__ void softmax_kernel(const float* __restrict__ scores,
                               float* __restrict__ weights_ws,
                               float* __restrict__ out_weights) {
    __shared__ float red[1024];
    int tid = threadIdx.x;
    float m = -3.402823466e38f;
    for (int i = tid; i < N_OBJ; i += 1024) m = fmaxf(m, scores[i]);
    red[tid] = m; __syncthreads();
    for (int s = 512; s > 0; s >>= 1) {
        if (tid < s) red[tid] = fmaxf(red[tid], red[tid + s]);
        __syncthreads();
    }
    float mx = red[0]; __syncthreads();
    float sum = 0.f;
    for (int i = tid; i < N_OBJ; i += 1024) sum += expf(scores[i] - mx);
    red[tid] = sum; __syncthreads();
    for (int s = 512; s > 0; s >>= 1) {
        if (tid < s) red[tid] += red[tid + s];
        __syncthreads();
    }
    float inv = 1.f / red[0];
    for (int i = tid; i < N_OBJ; i += 1024) {
        float w = expf(scores[i] - mx) * inv;
        weights_ws[i] = w;
        out_weights[i] = w;
    }
}

// Kernel G: related_feats[m,:] = weights[m] * obj[m,:] (obj likely L2-resident).
__global__ void scale_kernel(const float* __restrict__ obj,
                             const float* __restrict__ weights,
                             float* __restrict__ out) {
    int row = blockIdx.x;
    float w = weights[row];
    const float4* src = (const float4*)(obj + (size_t)row * D_OBJ);
    float4*       dst = (float4*)(out + (size_t)row * D_OBJ);
    for (int i = threadIdx.x; i < D_OBJ / 4; i += blockDim.x) {
        float4 v = src[i];
        dst[i] = make_float4(v.x * w, v.y * w, v.z * w, v.w * w);
    }
}

extern "C" void kernel_launch(void* const* d_in, const int* in_sizes, int n_in,
                              void* d_out, int out_size, void* d_ws, size_t ws_size,
                              hipStream_t stream) {
    const float* obj    = (const float*)d_in[0];
    const float* anchor = (const float*)d_in[1];
    const float* rel    = (const float*)d_in[2];
    const float* coords = (const float*)d_in[3];
    const float* Wa     = (const float*)d_in[4];
    const float* ba     = (const float*)d_in[5];
    const float* Wr     = (const float*)d_in[6];
    const float* br     = (const float*)d_in[7];
    const float* W1     = (const float*)d_in[8];
    const float* b1     = (const float*)d_in[9];
    const float* W2     = (const float*)d_in[10];
    const float* b2     = (const float*)d_in[11];
    float* out = (float*)d_out;

    float* ws         = (float*)d_ws;
    float* anchor_rep = ws;
    float* query      = ws + 4096;
    float* spatial    = ws + 8192;
    float* cvec       = ws + 20480;
    float* scores     = ws + 20608;
    float* weights    = ws + 24704;
    __bf16* W1bf      = (__bf16*)(ws + 28800);

    prep_kernel<<<16, 256, 0, stream>>>(anchor, coords, anchor_rep, spatial);
    query_gemv<<<512, 256, 0, stream>>>(Wa, ba, Wr, br, anchor_rep, rel, query);
    cvec_kernel<<<16, 256, 0, stream>>>(W1, b1, query, cvec);
    w1cvt_kernel<<<(H_MLP * D_OBJ) / 256, 256, 0, stream>>>(W1, W1bf);
    gemm_score_kernel<<<N_OBJ / 16, 256, 0, stream>>>(obj, W1bf, W1, cvec, spatial,
                                                      W2, b2, scores);
    softmax_kernel<<<1, 1024, 0, stream>>>(scores, weights, out + (size_t)N_OBJ * D_OBJ);
    scale_kernel<<<N_OBJ, 256, 0, stream>>>(obj, weights, out);
}
